// Contrast_39968965656798
// MI455X (gfx1250) — matrix operand order repaired and checked by reading, not compiled, not run on hardware
//
#include <hip/hip_runtime.h>
#include <hip/hip_bf16.h>

typedef _Float16 half_t;
typedef __attribute__((ext_vector_type(16))) _Float16 v16h;
typedef __attribute__((ext_vector_type(8)))  _Float16 v8h;
typedef __attribute__((ext_vector_type(4)))  _Float16 v4h;
typedef __attribute__((ext_vector_type(8)))  float    v8f;

#define NB   4096      // B
#define N2   8192      // 2B
#define DDIM 1024      // D
#define INV_T 2.0f     // 1 / 0.5

// ---------------------------------------------------------------------------
// Kernel 1: L2-normalize rows of [x_i; x_j] -> f16 z in workspace.
// One wave (32 lanes) per row; 8 rows per 256-thread block.
// ---------------------------------------------------------------------------
__global__ __launch_bounds__(256) void nrm_kernel(const float* __restrict__ xi,
                                                  const float* __restrict__ xj,
                                                  half_t* __restrict__ zh) {
    const int lane = threadIdx.x & 31;
    const int row  = blockIdx.x * 8 + (threadIdx.x >> 5);   // 0..8191
    const float* __restrict__ src =
        (row < NB) ? (xi + (size_t)row * DDIM) : (xj + (size_t)(row - NB) * DDIM);

    float4 vals[8];
    float ss = 0.0f;
#pragma unroll
    for (int i = 0; i < 8; ++i) {
        vals[i] = reinterpret_cast<const float4*>(src)[lane + 32 * i];
        ss += vals[i].x * vals[i].x + vals[i].y * vals[i].y +
              vals[i].z * vals[i].z + vals[i].w * vals[i].w;
    }
#pragma unroll
    for (int m = 16; m >= 1; m >>= 1) ss += __shfl_xor(ss, m, 32);

    const float inv = 1.0f / fmaxf(sqrtf(ss), 1e-12f);

    half_t* __restrict__ dst = zh + (size_t)row * DDIM;
#pragma unroll
    for (int i = 0; i < 8; ++i) {
        v4h h;
        h[0] = (half_t)(vals[i].x * inv);
        h[1] = (half_t)(vals[i].y * inv);
        h[2] = (half_t)(vals[i].z * inv);
        h[3] = (half_t)(vals[i].w * inv);
        reinterpret_cast<v4h*>(dst)[lane + 32 * i] = h;
    }
}

// ---------------------------------------------------------------------------
// Kernel 2: fused sim = z z^T tile GEMM (WMMA f16 -> f32) + exp row-sum.
//   grid = (8192/128, 8); block = 256 (8 waves). Wave w owns 16 rows
//   m0 = bx*128 + w*16. Block y-slice owns 1024 columns; per N-iteration a
//   wave produces two complete 16x16 sim tiles (full K), applies exp(2*sim),
//   drops the self-diagonal, records pos (col == row ^ 4096), accumulates
//   per-row partial denominators, then atomically adds them.
// ---------------------------------------------------------------------------
__global__ __launch_bounds__(256) void simclr_gemm(const half_t* __restrict__ zh,
                                                   float* __restrict__ denom,
                                                   float* __restrict__ pos) {
    const int lane = threadIdx.x & 31;
    const int wave = threadIdx.x >> 5;
    const int m0   = blockIdx.x * 128 + wave * 16;
    const int nBase = blockIdx.y * 1024;
    const int g = lane >> 4;          // half-wave: 0 or 1
    const int l = lane & 15;

    // A fragment source: row m0+l, K pattern {g*16 .. g*16+15}
    const half_t* __restrict__ aRow = zh + (size_t)(m0 + l) * DDIM;

    float rs[8];
#pragma unroll
    for (int v = 0; v < 8; ++v) rs[v] = 0.0f;

    for (int nt = 0; nt < 32; ++nt) {
        const int n0 = nBase + nt * 32;
        const half_t* __restrict__ b0Row = zh + (size_t)(n0 + l) * DDIM;       // cols n0..n0+15
        const half_t* __restrict__ b1Row = zh + (size_t)(n0 + 16 + l) * DDIM;  // cols n0+16..n0+31

        v8f c0 = {}, c1 = {};
        for (int k = 0; k < DDIM; k += 32) {
            v8h alo = *reinterpret_cast<const v8h*>(aRow + k + g * 16);
            v8h ahi = *reinterpret_cast<const v8h*>(aRow + k + g * 16 + 8);
            v16h a = __builtin_shufflevector(alo, ahi,
                0,1,2,3,4,5,6,7,8,9,10,11,12,13,14,15);

            v8h b0l = *reinterpret_cast<const v8h*>(b0Row + k + g * 16);
            v8h b0h = *reinterpret_cast<const v8h*>(b0Row + k + g * 16 + 8);
            v16h b0 = __builtin_shufflevector(b0l, b0h,
                0,1,2,3,4,5,6,7,8,9,10,11,12,13,14,15);

            v8h b1l = *reinterpret_cast<const v8h*>(b1Row + k + g * 16);
            v8h b1h = *reinterpret_cast<const v8h*>(b1Row + k + g * 16 + 8);
            v16h b1 = __builtin_shufflevector(b1l, b1h,
                0,1,2,3,4,5,6,7,8,9,10,11,12,13,14,15);

            c0 = __builtin_amdgcn_wmma_f32_16x16x32_f16(
                    false, a, false, b0, (short)0, c0, false, false);
            c1 = __builtin_amdgcn_wmma_f32_16x16x32_f16(
                    false, a, false, b1, (short)0, c1, false, false);
        }

        // Epilogue: c0[v] = sim[m0+v+8g][n0+l], c1[v] = sim[m0+v+8g][n0+16+l]
        const int col0 = n0 + l;
        const int col1 = n0 + 16 + l;
#pragma unroll
        for (int v = 0; v < 8; ++v) {
            const int r = m0 + v + 8 * g;
            const float s0 = c0[v];
            const float s1 = c1[v];
            if (r != col0) rs[v] += __expf(s0 * INV_T);   // exclude diagonal
            if (r != col1) rs[v] += __expf(s1 * INV_T);
            if ((r ^ NB) == col0) pos[r] = s0;            // positive pair sim
            if ((r ^ NB) == col1) pos[r] = s1;
        }
    }

    // Reduce partial row sums across the 16 lanes of each half-wave.
#pragma unroll
    for (int v = 0; v < 8; ++v) {
        float t = rs[v];
#pragma unroll
        for (int m = 1; m < 16; m <<= 1) t += __shfl_xor(t, m, 32);
        if (l == 0) atomicAdd(&denom[m0 + v + 8 * g], t);
    }
}

// ---------------------------------------------------------------------------
// Kernel 3: loss = mean_i( log(denom[i]) - pos[i]/T )
// ---------------------------------------------------------------------------
__global__ __launch_bounds__(256) void finalize_kernel(const float* __restrict__ denom,
                                                       const float* __restrict__ pos,
                                                       float* __restrict__ out) {
    __shared__ float sdata[256];
    float acc = 0.0f;
    for (int i = threadIdx.x; i < N2; i += 256)
        acc += logf(denom[i]) - pos[i] * INV_T;
    sdata[threadIdx.x] = acc;
    __syncthreads();
    for (int s = 128; s > 0; s >>= 1) {
        if (threadIdx.x < s) sdata[threadIdx.x] += sdata[threadIdx.x + s];
        __syncthreads();
    }
    if (threadIdx.x == 0) out[0] = sdata[0] / (float)N2;
}

// ---------------------------------------------------------------------------
extern "C" void kernel_launch(void* const* d_in, const int* in_sizes, int n_in,
                              void* d_out, int out_size, void* d_ws, size_t ws_size,
                              hipStream_t stream) {
    const float* xi = (const float*)d_in[0];
    const float* xj = (const float*)d_in[1];
    float* out = (float*)d_out;

    // Workspace layout: z_h (16 MB f16), denom (32 KB), pos (32 KB)
    half_t* zh   = (half_t*)d_ws;
    float* denom = (float*)((char*)d_ws + (size_t)N2 * DDIM * sizeof(half_t));
    float* pos   = denom + N2;

    hipMemsetAsync(denom, 0, 2 * N2 * sizeof(float), stream);

    nrm_kernel<<<N2 / 8, 256, 0, stream>>>(xi, xj, zh);

    dim3 grid(N2 / 128, 8);
    simclr_gemm<<<grid, 256, 0, stream>>>(zh, denom, pos);

    finalize_kernel<<<1, 256, 0, stream>>>(denom, pos, out);
}
